// ProdDecoder_67216238182747
// MI455X (gfx1250) — matrix operand order repaired and checked by reading, not compiled, not run on hardware
//
#include <hip/hip_runtime.h>
#include <hip/hip_bf16.h>

// ---------------------------------------------------------------------------
// ProdDecoder (tree-LSTM + dual global attention) for MI455X / gfx1250.
// Strategy: bf16 weights resident in 192MB L2, all GEMMs via
// v_wmma_f32_16x16x32_bf16 (wave32, 16x16 tiles, f32 accum).
// ---------------------------------------------------------------------------

#define NB 32
#define NT 256
#define NS 256
#define ND 512
#define NH 1024
#define NIN 2560   // 3*D + H
#define N4H 4096

typedef __bf16 v16bf __attribute__((ext_vector_type(16)));
typedef float  v8f   __attribute__((ext_vector_type(8)));

union BF16x16 { uint4 u[2]; v16bf v; };

__device__ __forceinline__ unsigned short f2bf(float f) {
    union { float f; unsigned int u; } v; v.f = f;
    unsigned int u = v.u;
    unsigned int r = (u + 0x7FFFu + ((u >> 16) & 1u)) >> 16;   // RNE
    return (unsigned short)r;
}
__device__ __forceinline__ float bf2f(unsigned short h) {
    union { unsigned int u; float f; } v; v.u = ((unsigned int)h) << 16;
    return v.f;
}
__device__ __forceinline__ float sigmoidf(float x) {
    return 1.0f / (1.0f + __expf(-x));
}

// ---------------------------------------------------------------------------
// Generic skinny GEMM:  D[m,n] = act( (accum? D : 0) + A[m,:]·Bw[n,:] + bias[n] )
//   A  : [32, K]  row-major bf16 (lda = K)
//   Bw : [N, K]   row-major bf16 (ldb = K)  (i.e. computes A @ Bw^T)
// One wave per (mtile, ntile): blockDim=64 (2 waves = 2 m-tiles), grid = N/16.
// A-tile / B-tile lane layouts follow CDNA5 ISA 7.12.2 bf16 tables; both map
// to contiguous 16B global loads.
// ---------------------------------------------------------------------------
__global__ void gemm_wmma_bf16(const unsigned short* __restrict__ A, int lda,
                               const unsigned short* __restrict__ Bw, int ldb,
                               const float* __restrict__ bias,
                               float* __restrict__ D, long long dRowStride,
                               long long dColOff,
                               unsigned short* __restrict__ Dbf, int dbfStride,
                               int K, int accum, int act_tanh)
{
    const int mtile = threadIdx.x >> 5;       // wave id in block: 0..1
    const int lane  = threadIdx.x & 31;
    const int ntile = blockIdx.x;
    const int nloc  = lane & 15;
    const int half  = lane >> 4;

    const unsigned short* abase = A  + (size_t)(mtile * 16 + nloc) * lda + half * 8;
    const unsigned short* bbase = Bw + (size_t)(ntile * 16 + nloc) * ldb + half * 8;

    v8f acc = {};
    for (int k0 = 0; k0 < K; k0 += 32) {
        BF16x16 a, b;
        a.u[0] = *(const uint4*)(abase + k0);        // V0..V3: K = k0+half*8 .. +7
        a.u[1] = *(const uint4*)(abase + k0 + 16);   // V4..V7: K = k0+16+half*8 ..
        b.u[0] = *(const uint4*)(bbase + k0);        // K = k0+half*8 .. +7
        b.u[1] = *(const uint4*)(bbase + k0 + 16);   // K = k0+16+half*8 .. +7
        acc = __builtin_amdgcn_wmma_f32_16x16x32_bf16(
                  false, a.v, false, b.v, (short)0, acc, false, false);
    }

    const int ng    = ntile * 16 + nloc;
    const int mbase = mtile * 16 + half * 8;
    const float bv  = bias ? bias[ng] : 0.0f;
#pragma unroll
    for (int r = 0; r < 8; ++r) {
        long long off = (long long)(mbase + r) * dRowStride + dColOff + ng;
        float v = acc[r] + bv;
        if (accum)    v += D[off];
        if (act_tanh) v = tanhf(v);
        D[off] = v;
        if (Dbf) Dbf[(size_t)(mbase + r) * dbfStride + ng] = f2bf(v);
    }
}

// ---------------------------------------------------------------------------
// One-time prep kernels
// ---------------------------------------------------------------------------
__global__ void convert_bf16_kernel(const float* __restrict__ src,
                                    unsigned short* __restrict__ dst, size_t n) {
    size_t i = (size_t)blockIdx.x * blockDim.x + threadIdx.x;
    if (i < n) dst[i] = f2bf(src[i]);
}

__global__ void zero_us_kernel(unsigned short* __restrict__ dst, size_t n) {
    size_t i = (size_t)blockIdx.x * blockDim.x + threadIdx.x;
    if (i < n) dst[i] = 0;
}

// Gather all embeddings for every (b,t): Xemb[b,t,0:512]=nt_emb[nt],
// [512:1024]=rule_emb[prev], [1024:1536]=rule_emb[parent]   (bf16)
__global__ void embed_kernel(const int* __restrict__ nt,
                             const int* __restrict__ prev,
                             const int* __restrict__ par,
                             const float* __restrict__ nt_emb,
                             const float* __restrict__ rule_emb,
                             unsigned short* __restrict__ Xemb) {
    size_t i = (size_t)blockIdx.x * blockDim.x + threadIdx.x;
    if (i >= (size_t)NB * NT * 3 * ND) return;
    int    d  = (int)(i % (3 * ND));
    size_t bt = i / (3 * ND);
    float v;
    if (d < ND)            v = nt_emb[(size_t)nt[bt]   * ND + d];
    else if (d < 2 * ND)   v = rule_emb[(size_t)prev[bt] * ND + (d - ND)];
    else                   v = rule_emb[(size_t)par[bt]  * ND + (d - 2 * ND)];
    Xemb[i] = f2bf(v);
}

__global__ void init_state_kernel(const float* __restrict__ h0,
                                  const float* __restrict__ c0,
                                  float* __restrict__ h, float* __restrict__ c,
                                  unsigned short* __restrict__ hbf) {
    int i = blockIdx.x * blockDim.x + threadIdx.x;   // 32*1024
    h[i] = h0[i]; c[i] = c0[i]; hbf[i] = f2bf(h0[i]);
}

__global__ void bias_kernel(const float* __restrict__ b_ih,
                            const float* __restrict__ b_hh,
                            float* __restrict__ bias4h) {
    int i = blockIdx.x * blockDim.x + threadIdx.x;   // 4096
    bias4h[i] = b_ih[i] + b_hh[i];
}

// ---------------------------------------------------------------------------
// Per-step kernels
// ---------------------------------------------------------------------------
// xt[b, 0:1536] = Xemb[b,t,:]; xt[b,1536:2560] = parent state from buf
__global__ void prep_x_kernel(const unsigned short* __restrict__ Xemb,
                              const unsigned short* __restrict__ bufbf,
                              const int* __restrict__ parent_t, int t,
                              unsigned short* __restrict__ xt) {
    int i = blockIdx.x * blockDim.x + threadIdx.x;   // 32*2560
    int b = i / NIN, k = i - b * NIN;
    unsigned short v;
    if (k < 3 * ND) {
        v = Xemb[((size_t)b * NT + t) * (3 * ND) + k];
    } else {
        int pidx = parent_t[b * NT + t];
        int slot = (pidx < t) ? (pidx + 1) : 0;
        v = bufbf[((size_t)b * (NT + 1) + slot) * NH + (k - 3 * ND)];
    }
    xt[i] = v;
}

// LSTM cell (torch gate order i,f,g,o) + scatter h_new into parent buffer
__global__ void lstm_kernel(const float* __restrict__ gates,
                            float* __restrict__ c, float* __restrict__ h,
                            unsigned short* __restrict__ hbf,
                            unsigned short* __restrict__ bufbf, int t) {
    int i = blockIdx.x * blockDim.x + threadIdx.x;   // 32*1024
    int b = i >> 10, hh = i & (NH - 1);
    const float* g = gates + (size_t)b * N4H;
    float ig = sigmoidf(g[hh]);
    float fg = sigmoidf(g[NH + hh]);
    float gg = tanhf   (g[2 * NH + hh]);
    float og = sigmoidf(g[3 * NH + hh]);
    float cn = fg * c[i] + ig * gg;
    float hn = og * tanhf(cn);
    c[i] = cn; h[i] = hn;
    unsigned short hb = f2bf(hn);
    hbf[i] = hb;
    bufbf[((size_t)b * (NT + 1) + t + 1) * NH + hh] = hb;
}

// scores[b,s] = q[b,:]·ctx[b,s,:]  (masked), one wave per (b,s)
__global__ void attn_scores_kernel(const float* __restrict__ q,
                                   const unsigned short* __restrict__ ctx,
                                   const int* __restrict__ lens,
                                   float* __restrict__ scores) {
    int gw   = (blockIdx.x * blockDim.x + threadIdx.x) >> 5;   // 0..8191
    int lane = threadIdx.x & 31;
    int b = gw >> 8, s = gw & (NS - 1);
    const float*          qb = q   + (size_t)b * NH;
    const unsigned short* cb = ctx + ((size_t)b * NS + s) * NH;
    float acc = 0.0f;
    for (int hh = lane; hh < NH; hh += 32)
        acc += qb[hh] * bf2f(cb[hh]);
#pragma unroll
    for (int m = 16; m >= 1; m >>= 1)
        acc += __shfl_xor(acc, m, 32);
    if (lane == 0)
        scores[b * NS + s] = (s < lens[b]) ? acc : -3.0e38f;
}

// row softmax over S=256 (one block per b); writes probs to ws and to d_out
__global__ void softmax_kernel(const float* __restrict__ scores,
                               float* __restrict__ p_ws,
                               float* __restrict__ p_out /* + t*S folded in */) {
    __shared__ float red[8];
    int b = blockIdx.x, s = threadIdx.x;
    float x = scores[b * NS + s];
    float m = x;
#pragma unroll
    for (int d = 16; d >= 1; d >>= 1) m = fmaxf(m, __shfl_xor(m, d, 32));
    if ((s & 31) == 0) red[s >> 5] = m;
    __syncthreads();
    float mm = red[0];
#pragma unroll
    for (int i = 1; i < 8; ++i) mm = fmaxf(mm, red[i]);
    float e = __expf(x - mm);
    float ws_ = e;
#pragma unroll
    for (int d = 16; d >= 1; d >>= 1) ws_ += __shfl_xor(ws_, d, 32);
    __syncthreads();
    if ((s & 31) == 0) red[s >> 5] = ws_;
    __syncthreads();
    float tot = 0.0f;
#pragma unroll
    for (int i = 0; i < 8; ++i) tot += red[i];
    float p = e / tot;
    p_ws[b * NS + s] = p;
    p_out[(size_t)b * NT * NS + s] = p;
}

// c_vec[b,h] = sum_s p[b,s]*ctx[b,s,h]; pack cat = [c_vec, q] as bf16
__global__ void cvec_cat_kernel(const float* __restrict__ p,
                                const unsigned short* __restrict__ ctx,
                                const unsigned short* __restrict__ qbf,
                                unsigned short* __restrict__ cat) {
    int i = blockIdx.x * blockDim.x + threadIdx.x;   // 32*1024
    int b = i >> 10, hh = i & (NH - 1);
    const unsigned short* cb = ctx + (size_t)b * NS * NH + hh;
    const float*          pb = p   + b * NS;
    float acc = 0.0f;
    for (int s = 0; s < NS; ++s)
        acc += pb[s] * bf2f(cb[(size_t)s * NH]);
    cat[(size_t)b * (2 * NH) + hh]      = f2bf(acc);
    cat[(size_t)b * (2 * NH) + NH + hh] = qbf[i];
}

// ---------------------------------------------------------------------------
extern "C" void kernel_launch(void* const* d_in, const int* in_sizes, int n_in,
                              void* d_out, int out_size, void* d_ws, size_t ws_size,
                              hipStream_t stream) {
    (void)in_sizes; (void)n_in; (void)out_size; (void)ws_size;
    const int*   nt        = (const int*)d_in[0];
    const int*   prev      = (const int*)d_in[1];
    const int*   par       = (const int*)d_in[2];
    const int*   parent_t  = (const int*)d_in[3];
    const float* context   = (const float*)d_in[4];
    const int*   lens      = (const int*)d_in[5];
    const float* h0        = (const float*)d_in[6];
    const float* c0        = (const float*)d_in[7];
    const float* nt_emb    = (const float*)d_in[8];
    const float* rule_emb  = (const float*)d_in[9];
    const float* W_ih      = (const float*)d_in[10];
    const float* W_hh      = (const float*)d_in[11];
    const float* b_ih      = (const float*)d_in[12];
    const float* b_hh      = (const float*)d_in[13];
    const float* Wa_in     = (const float*)d_in[14];
    const float* Wa_out    = (const float*)d_in[15];
    const float* Wc_in     = (const float*)d_in[16];
    // d_in[17] = Wc_out: unused — reference discards 2nd attention output.

    float* out_o = (float*)d_out;                                   // [B,T,H]
    float* out_a = out_o + (size_t)NB * NT * NH;                    // [B,T,S]
    float* out_c = out_a + (size_t)NB * NT * NS;                    // [B,T,S]

    // ---- workspace carve-up (bf16 weights stay L2-resident) ----
    char* ws = (char*)d_ws;
    size_t off = 0;
    auto aus = [&](size_t n) { size_t o = off; off += ((n * 2 + 255) / 256) * 256;
                               return (unsigned short*)(ws + o); };
    auto af  = [&](size_t n) { size_t o = off; off += ((n * 4 + 255) / 256) * 256;
                               return (float*)(ws + o); };
    unsigned short* Wih_b  = aus((size_t)N4H * NIN);
    unsigned short* Whh_b  = aus((size_t)N4H * NH);
    unsigned short* Wain_b = aus((size_t)NH * NH);
    unsigned short* Waout_b= aus((size_t)NH * 2 * NH);
    unsigned short* Wcin_b = aus((size_t)NH * NH);
    unsigned short* ctx_b  = aus((size_t)NB * NS * NH);
    unsigned short* Xemb   = aus((size_t)NB * NT * 3 * ND);
    unsigned short* bufbf  = aus((size_t)NB * (NT + 1) * NH);
    unsigned short* xt     = aus((size_t)NB * NIN);
    unsigned short* hbf    = aus((size_t)NB * NH);
    unsigned short* aoutbf = aus((size_t)NB * NH);
    unsigned short* catbf  = aus((size_t)NB * 2 * NH);
    float* hbuf   = af((size_t)NB * NH);
    float* cbuf   = af((size_t)NB * NH);
    float* gates  = af((size_t)NB * N4H);
    float* qbuf   = af((size_t)NB * NH);
    float* scbuf  = af((size_t)NB * NS);
    float* pbuf   = af((size_t)NB * NS);
    float* bias4h = af((size_t)N4H);

    auto cdiv = [](size_t a, size_t b) { return (a + b - 1) / b; };

    // ---- one-time prep ----
    convert_bf16_kernel<<<cdiv((size_t)N4H * NIN, 256), 256, 0, stream>>>(W_ih,  Wih_b,  (size_t)N4H * NIN);
    convert_bf16_kernel<<<cdiv((size_t)N4H * NH,  256), 256, 0, stream>>>(W_hh,  Whh_b,  (size_t)N4H * NH);
    convert_bf16_kernel<<<cdiv((size_t)NH  * NH,  256), 256, 0, stream>>>(Wa_in, Wain_b, (size_t)NH * NH);
    convert_bf16_kernel<<<cdiv((size_t)NH*2*NH,   256), 256, 0, stream>>>(Wa_out,Waout_b,(size_t)NH * 2 * NH);
    convert_bf16_kernel<<<cdiv((size_t)NH  * NH,  256), 256, 0, stream>>>(Wc_in, Wcin_b, (size_t)NH * NH);
    convert_bf16_kernel<<<cdiv((size_t)NB*NS*NH,  256), 256, 0, stream>>>(context, ctx_b, (size_t)NB * NS * NH);
    embed_kernel<<<cdiv((size_t)NB * NT * 3 * ND, 256), 256, 0, stream>>>(nt, prev, par, nt_emb, rule_emb, Xemb);
    zero_us_kernel<<<cdiv((size_t)NB * (NT + 1) * NH, 256), 256, 0, stream>>>(bufbf, (size_t)NB * (NT + 1) * NH);
    init_state_kernel<<<NB * NH / 256, 256, 0, stream>>>(h0, c0, hbuf, cbuf, hbf);
    bias_kernel<<<N4H / 256, 256, 0, stream>>>(b_ih, b_hh, bias4h);

    // ---- sequential decode: 256 steps ----
    for (int t = 0; t < NT; ++t) {
        // x_t = [embeddings(t), parent_state]
        prep_x_kernel<<<NB * NIN / 256, 256, 0, stream>>>(Xemb, bufbf, parent_t, t, xt);
        // gates = x_t @ W_ih^T + h @ W_hh^T + bias
        gemm_wmma_bf16<<<N4H / 16, 64, 0, stream>>>(xt, NIN, Wih_b, NIN, bias4h,
            gates, N4H, 0, nullptr, 0, NIN, /*accum*/0, /*tanh*/0);
        gemm_wmma_bf16<<<N4H / 16, 64, 0, stream>>>(hbf, NH, Whh_b, NH, nullptr,
            gates, N4H, 0, nullptr, 0, NH, /*accum*/1, /*tanh*/0);
        // LSTM cell; h_new -> hbf + buffer slot t+1
        lstm_kernel<<<NB * NH / 256, 256, 0, stream>>>(gates, cbuf, hbuf, hbf, bufbf, t);

        // attention 1: q = h_new
        gemm_wmma_bf16<<<NH / 16, 64, 0, stream>>>(hbf, NH, Wain_b, NH, nullptr,
            qbuf, NH, 0, nullptr, 0, NH, 0, 0);
        attn_scores_kernel<<<NB * NS / 8, 256, 0, stream>>>(qbuf, ctx_b, lens, scbuf);
        softmax_kernel<<<NB, NS, 0, stream>>>(scbuf, pbuf, out_a + (size_t)t * NS);
        cvec_cat_kernel<<<NB * NH / 256, 256, 0, stream>>>(pbuf, ctx_b, hbf, catbf);
        // a_out = tanh([c_vec, q] @ Wa_out^T) -> d_out[:,t,:] and bf16 copy
        gemm_wmma_bf16<<<NH / 16, 64, 0, stream>>>(catbf, 2 * NH, Waout_b, 2 * NH, nullptr,
            out_o, (long long)NT * NH, (long long)t * NH, aoutbf, NH, 2 * NH, 0, /*tanh*/1);

        // attention 2 (copy scores only): q = a_out
        gemm_wmma_bf16<<<NH / 16, 64, 0, stream>>>(aoutbf, NH, Wcin_b, NH, nullptr,
            qbuf, NH, 0, nullptr, 0, NH, 0, 0);
        attn_scores_kernel<<<NB * NS / 8, 256, 0, stream>>>(qbuf, ctx_b, lens, scbuf);
        softmax_kernel<<<NB, NS, 0, stream>>>(scbuf, pbuf, out_c + (size_t)t * NS);
    }
}
